// ANsDiscovery_54984171323424
// MI455X (gfx1250) — matrix operand order repaired and checked by reading, not compiled, not run on hardware
//
#include <hip/hip_runtime.h>
#include <hip/hip_bf16.h>

typedef _Float16 h8  __attribute__((ext_vector_type(8)));
typedef _Float16 h16 __attribute__((ext_vector_type(16)));
typedef float    f8  __attribute__((ext_vector_type(8)));

#define N_SAMPLES 20000
#define D_FEAT    128
#define ANS_NUM   5000          // round == 1 (output layout fixes this)
#define ANS_SIZE  10
#define INV_TEMP  10.0f
#define MSHIFT    10.5f         // fixed logsumexp shift: z = sim*10 <= ~10 for unit vectors
#define COL_TILES (N_SAMPLES / 16)          // 1250 (exact)
#define ROW_PAIRS (COL_TILES / 2)           // 625 (exact)
#define ROW_TILES_A ((ANS_NUM + 15) / 16)   // 313
#define SORT_N    32768

// d_out layout (floats), reference return order:
#define OUT_ENT   0        // entropy            [20000]
#define OUT_ANC   20000    // anchor_indexes     [5000]
#define OUT_INST  25000    // instance_indexes   [15000]
#define OUT_POS   40000    // position           [20000]
#define OUT_NBR   60000    // neighbours         [50000]
#define OUT_CONS  110000   // consistency        [1]

union H16U { h16 v; h8 h[2]; };

// A-matrix 16x32 f16 fragment (ISA 7.12.2): lane<16 -> K{0..7,16..23}, lane>=16 -> K{8..15,24..31}
__device__ inline h16 load_a_frag(const _Float16* rowp, int kk, int lane) {
    const _Float16* p = rowp + kk * 32 + ((lane & 16) ? 8 : 0);
    H16U u;
    u.h[0] = *(const h8*)(p);
    u.h[1] = *(const h8*)(p + 16);
    return u.v;
}
// B-matrix 32x16 f16 fragment: lane<16 -> K{0..7,16..23}, lane>=16 -> K{8..15,24..31}
__device__ inline h16 load_b_frag(const _Float16* colp, int kk, int lane) {
    const _Float16* p = colp + kk * 32 + ((lane & 16) ? 8 : 0);
    H16U u;
    u.h[0] = *(const h8*)(p);
    u.h[1] = *(const h8*)(p + 16);
    return u.v;
}

// ---------------- kernel 0: f32 -> f16 feature copy ----------------
__global__ void k_cvt_f16(const float* __restrict__ src, _Float16* __restrict__ dst, int n) {
    int i = blockIdx.x * blockDim.x + threadIdx.x;
    int stride = gridDim.x * blockDim.x;
    for (; i < n; i += stride) dst[i] = (_Float16)src[i];
}

// ---------------- kernel 1: fused GEMM + fixed-shift softmax entropy ----------------
// One wave handles TWO 16-row tiles (32 rows) so every B fragment is reused twice
// (halves L2 panel traffic). Unit-norm features => z = sim*10 in [-10.01, 10.01],
// so a fixed shift M=10.5 is a safe logsumexp shift: no running max needed.
//   s = sum exp(z-M), t = sum z*exp(z-M);  H = M + log(s) - t/s
__global__ __launch_bounds__(256) void k_entropy(const _Float16* __restrict__ feat,
                                                 float* __restrict__ out_ent) {
    const int lane = threadIdx.x & 31;
    const int wave = threadIdx.x >> 5;
    const int pair = blockIdx.x * 8 + wave;
    if (pair >= ROW_PAIRS) return;              // uniform per wave
    const int rowBase0 = pair * 32;
    const int rowBase1 = rowBase0 + 16;
    const _Float16* rowp0 = feat + (size_t)(rowBase0 + (lane & 15)) * D_FEAT;
    const _Float16* rowp1 = feat + (size_t)(rowBase1 + (lane & 15)) * D_FEAT;

    h16 a0[4], a1[4];
#pragma unroll
    for (int kk = 0; kk < 4; ++kk) {
        a0[kk] = load_a_frag(rowp0, kk, lane);
        a1[kk] = load_a_frag(rowp1, kk, lane);
    }

    float s0[8], t0[8], s1[8], t1[8];
#pragma unroll
    for (int v = 0; v < 8; ++v) { s0[v] = 0.f; t0[v] = 0.f; s1[v] = 0.f; t1[v] = 0.f; }

    for (int ct = 0; ct < COL_TILES; ++ct) {
        const _Float16* colp = feat + (size_t)(ct * 16 + (lane & 15)) * D_FEAT;
        h16 b[4];
#pragma unroll
        for (int kk = 0; kk < 4; ++kk) b[kk] = load_b_frag(colp, kk, lane);

        f8 acc0 = {}, acc1 = {};
#pragma unroll
        for (int kk = 0; kk < 4; ++kk) {
            acc0 = __builtin_amdgcn_wmma_f32_16x16x32_f16(false, a0[kk], false, b[kk], (short)0, acc0, false, false);
            acc1 = __builtin_amdgcn_wmma_f32_16x16x32_f16(false, a1[kk], false, b[kk], (short)0, acc1, false, false);
        }
#pragma unroll
        for (int v = 0; v < 8; ++v) {
            float z0 = acc0[v] * INV_TEMP;
            float e0 = __expf(z0 - MSHIFT);
            s0[v] += e0;
            t0[v] = __fmaf_rn(z0, e0, t0[v]);
            float z1 = acc1[v] * INV_TEMP;
            float e1 = __expf(z1 - MSHIFT);
            s1[v] += e1;
            t1[v] = __fmaf_rn(z1, e1, t1[v]);
        }
    }
    // plain shuffle-add across the 16-lane group (no rescale needed: fixed shift)
#pragma unroll
    for (int off = 1; off < 16; off <<= 1) {
#pragma unroll
        for (int v = 0; v < 8; ++v) {
            s0[v] += __shfl_xor(s0[v], off, 32);
            t0[v] += __shfl_xor(t0[v], off, 32);
            s1[v] += __shfl_xor(s1[v], off, 32);
            t1[v] += __shfl_xor(t1[v], off, 32);
        }
    }
    if ((lane & 15) == 0) {
        int half = lane >> 4;
#pragma unroll
        for (int v = 0; v < 8; ++v) {
            int r0 = rowBase0 + v + half * 8;
            int r1 = rowBase1 + v + half * 8;
            out_ent[r0] = MSHIFT + __logf(s0[v]) - t0[v] / s0[v];
            out_ent[r1] = MSHIFT + __logf(s1[v]) - t1[v] / s1[v];
        }
    }
}

// ---------------- kernel 2: full sort in 256KB LDS + selection bookkeeping ----------------
// CDNA5 320KB LDS lets one WGP hold all 32768 (entropy,index) 64-bit keys.
__global__ __launch_bounds__(1024) void k_sort_select(const float* __restrict__ ent,
                                                      int* __restrict__ anchorIdx,
                                                      int* __restrict__ isAnchor,
                                                      int* __restrict__ anchorPos,
                                                      float* __restrict__ dout) {
    __shared__ unsigned long long keys[SORT_N];   // 262144 B
    __shared__ int ssum[1024];
    const int tid = threadIdx.x;

    for (int i = tid; i < SORT_N; i += 1024) {
        if (i < N_SAMPLES) {
            unsigned int b = __float_as_uint(ent[i]);
            b ^= (b & 0x80000000u) ? 0xFFFFFFFFu : 0x80000000u;   // total order on floats
            keys[i] = ((unsigned long long)b << 32) | (unsigned int)i;
        } else {
            keys[i] = ~0ull;
        }
    }
    for (int i = tid; i < N_SAMPLES; i += 1024) isAnchor[i] = 0;
    __syncthreads();

    for (int k = 2; k <= SORT_N; k <<= 1) {
        for (int j = k >> 1; j > 0; j >>= 1) {
            for (int i = tid; i < SORT_N; i += 1024) {
                int p = i ^ j;
                if (p > i) {
                    unsigned long long a = keys[i], b = keys[p];
                    bool up = ((i & k) == 0);
                    if ((a > b) == up) { keys[i] = b; keys[p] = a; }
                }
            }
            __syncthreads();
        }
    }
    // ascending (entropy,index) == jax top_k(-entropy) order with its tie-break
    for (int i = tid; i < ANS_NUM; i += 1024) {
        int idx = (int)(unsigned int)(keys[i] & 0xFFFFFFFFull);
        anchorIdx[i] = idx;
        dout[OUT_ANC + i] = (float)idx;
        isAnchor[idx] = 1;
        anchorPos[idx] = i;
    }
    __syncthreads();

    // block-wide scan of non-anchor counts -> position / instance_indexes
    const int CH = 20;                       // 1024*20 >= 20000
    int base = tid * CH;
    int cnt = 0;
    for (int e = 0; e < CH; ++e) { int j = base + e; if (j < N_SAMPLES) cnt += (isAnchor[j] == 0); }
    ssum[tid] = cnt;
    __syncthreads();
    for (int off = 1; off < 1024; off <<= 1) {
        int v = (tid >= off) ? ssum[tid - off] : 0;
        __syncthreads();
        ssum[tid] += v;
        __syncthreads();
    }
    int run = ssum[tid] - cnt;               // exclusive prefix
    for (int e = 0; e < CH; ++e) {
        int j = base + e;
        if (j < N_SAMPLES) {
            if (isAnchor[j]) {
                dout[OUT_POS + j] = (float)anchorPos[j];
            } else {
                run++;
                dout[OUT_POS + j] = (float)(-run);
                dout[OUT_INST + (run - 1)] = (float)j;
            }
        }
    }
}

// ---------------- kernel 3: anchor GEMM + streaming top-10 neighbours ----------------
__global__ __launch_bounds__(256) void k_neighbours(const _Float16* __restrict__ feat,
                                                    const int* __restrict__ anchorIdx,
                                                    int* __restrict__ nbrWs,
                                                    float* __restrict__ dout) {
    __shared__ float tile[8][16][17];
    const int lane = threadIdx.x & 31;
    const int wave = threadIdx.x >> 5;
    const int rowTile = blockIdx.x * 8 + wave;
    if (rowTile >= ROW_TILES_A) return;
    const int rowBase = rowTile * 16;
    int myRow = rowBase + (lane & 15);
    int aIdx = anchorIdx[myRow < ANS_NUM ? myRow : (ANS_NUM - 1)];
    const _Float16* rowp = feat + (size_t)aIdx * D_FEAT;

    h16 a0 = load_a_frag(rowp, 0, lane);
    h16 a1 = load_a_frag(rowp, 1, lane);
    h16 a2 = load_a_frag(rowp, 2, lane);
    h16 a3 = load_a_frag(rowp, 3, lane);

    float tv[10]; int ti[10];
#pragma unroll
    for (int k = 0; k < 10; ++k) { tv[k] = -2.0f; ti[k] = 0x7FFFFFFF; }

    const int l  = lane & 15;
    const int j0 = (lane & 16) ? 8 : 0;      // lane l and l+16 split the 16 columns of row l

    for (int ct = 0; ct < COL_TILES; ++ct) {
        const _Float16* colp = feat + (size_t)(ct * 16 + (lane & 15)) * D_FEAT;
        f8 acc = {};
        acc = __builtin_amdgcn_wmma_f32_16x16x32_f16(false, a0, false, load_b_frag(colp, 0, lane), (short)0, acc, false, false);
        acc = __builtin_amdgcn_wmma_f32_16x16x32_f16(false, a1, false, load_b_frag(colp, 1, lane), (short)0, acc, false, false);
        acc = __builtin_amdgcn_wmma_f32_16x16x32_f16(false, a2, false, load_b_frag(colp, 2, lane), (short)0, acc, false, false);
        acc = __builtin_amdgcn_wmma_f32_16x16x32_f16(false, a3, false, load_b_frag(colp, 3, lane), (short)0, acc, false, false);
#pragma unroll
        for (int v = 0; v < 8; ++v)
            tile[wave][v + ((lane & 16) ? 8 : 0)][lane & 15] = acc[v];
        // same-wave LDS RAW: in-order DS + compiler s_wait_dscnt; no barrier needed
#pragma unroll
        for (int j = 0; j < 8; ++j) {
            float val = tile[wave][l][j0 + j];
            int col = ct * 16 + j0 + j;
            if (col == aIdx) val = -1.0f;    // self mask (reference sets sims[i,self] = -1)
            if (val > tv[9]) {               // stable insert (strict >): lower col wins ties
                tv[9] = val; ti[9] = col;
#pragma unroll
                for (int k = 9; k > 0; --k) {
                    if (tv[k] > tv[k - 1]) {
                        float tf = tv[k]; tv[k] = tv[k - 1]; tv[k - 1] = tf;
                        int   tt = ti[k]; ti[k] = ti[k - 1]; ti[k - 1] = tt;
                    }
                }
            }
        }
    }
    // merge the two half-column top-10 lists (lane l with lane l+16)
    float pv[10]; int pi[10];
#pragma unroll
    for (int k = 0; k < 10; ++k) {
        pv[k] = __shfl_xor(tv[k], 16, 32);
        pi[k] = __shfl_xor(ti[k], 16, 32);
    }
    if ((lane < 16) && (rowBase + lane) < ANS_NUM) {
        int r = rowBase + lane;
        int ia = 0, ib = 0;
#pragma unroll
        for (int k = 0; k < 10; ++k) {
            float va = tv[ia], vb = pv[ib];
            int   ca = ti[ia], cb = pi[ib];
            bool pickA = (va > vb) || (va == vb && ca < cb);
            int idx = pickA ? ca : cb;
            if (pickA) ia++; else ib++;
            nbrWs[r * ANS_SIZE + k] = idx;
            dout[OUT_NBR + r * ANS_SIZE + k] = (float)idx;
        }
    }
}

// ---------------- kernel 4: label consistency (integer reduction, deterministic) ----------------
__global__ __launch_bounds__(256) void k_consistency(const int* __restrict__ labels,
                                                     const int* __restrict__ anchorIdx,
                                                     const int* __restrict__ nbrWs,
                                                     float* __restrict__ dout) {
    __shared__ int red[256];
    int tid = threadIdx.x;
    int cnt = 0;
    for (int i = tid; i < ANS_NUM * ANS_SIZE; i += 256) {
        int a = anchorIdx[i / ANS_SIZE];
        cnt += (labels[a] == labels[nbrWs[i]]) ? 1 : 0;
    }
    red[tid] = cnt;
    __syncthreads();
    for (int off = 128; off > 0; off >>= 1) {
        if (tid < off) red[tid] += red[tid + off];
        __syncthreads();
    }
    if (tid == 0) dout[OUT_CONS] = (float)red[0] / (float)(ANS_NUM * ANS_SIZE);
}

extern "C" void kernel_launch(void* const* d_in, const int* in_sizes, int n_in,
                              void* d_out, int out_size, void* d_ws, size_t ws_size,
                              hipStream_t stream) {
    (void)in_sizes; (void)n_in; (void)out_size; (void)ws_size;
    const float* feats  = (const float*)d_in[0];
    const int*   labels = (const int*)d_in[1];
    // d_in[2] = round; output layout is fixed for round==1
    float* out = (float*)d_out;

    char* ws = (char*)d_ws;
    _Float16* feat16   = (_Float16*)(ws);                 //  5,120,000 B
    int* anchorIdx     = (int*)(ws + 5120000);            //     20,000 B
    int* isAnchor      = (int*)(ws + 5140000);            //     80,000 B
    int* anchorPos     = (int*)(ws + 5220000);            //     80,000 B
    int* nbrWs         = (int*)(ws + 5300000);            //    200,000 B

    k_cvt_f16<<<320, 256, 0, stream>>>(feats, feat16, N_SAMPLES * D_FEAT);
    k_entropy<<<(ROW_PAIRS + 7) / 8, 256, 0, stream>>>(feat16, out + OUT_ENT);
    k_sort_select<<<1, 1024, 0, stream>>>(out + OUT_ENT, anchorIdx, isAnchor, anchorPos, out);
    k_neighbours<<<(ROW_TILES_A + 7) / 8, 256, 0, stream>>>(feat16, anchorIdx, nbrWs, out);
    k_consistency<<<1, 256, 0, stream>>>(labels, anchorIdx, nbrWs, out);
}
